// YOURLSTM_15324443312435
// MI455X (gfx1250) — matrix operand order repaired and checked by reading, not compiled, not run on hardware
//
#include <hip/hip_runtime.h>
#include <stdint.h>

#define NLAYERS 4
#define BATCH   4096
#define HID     1024
#define KDIM    2048        // input(1024) + hidden(1024) fused along K
#define NDIM    4096        // 4 * HID (gate-interleaved columns: c = 4*h + gate)

#define BM 128
#define BN 128
#define BK 32
#define LDT 40              // LDS row stride (bf16 elems); 80B rows keep b128 alignment
#define NKSTEP (KDIM / BK)  // 64

typedef __attribute__((ext_vector_type(16))) __bf16 bf16x16;
typedef __attribute__((ext_vector_type(8)))  float  f32x8;

union FragU { uint4 q[2]; bf16x16 v; };

__device__ __forceinline__ uint16_t f2bf(float f) {
  union { float f; uint32_t u; } cv; cv.f = f;
  uint32_t u = cv.u;
  uint32_t r = u + 0x7FFFu + ((u >> 16) & 1u);   // round-to-nearest-even
  return (uint16_t)(r >> 16);
}

__device__ __forceinline__ float sigf(float x) {
  return 1.0f / (1.0f + __expf(-x));
}

// ---- Pack [W_ih ; W_hh] fp32 -> bf16, transposed to [c][k], gate-interleaved columns.
__global__ void lstm_pack_w(const float* __restrict__ W_ih, const float* __restrict__ W_hh,
                            uint16_t* __restrict__ Wp) {
  long idx = (long)blockIdx.x * blockDim.x + threadIdx.x;   // < NLAYERS*NDIM*(KDIM/4)
  int k4 = (int)(idx & 511) << 2;          // 0..2044 step 4
  int c  = (int)(idx >> 9) & 4095;
  int l  = (int)(idx >> 21);
  int h = c >> 2, g = c & 3;
  int srow = g * HID + h;
  const float* src = (k4 < 1024)
      ? (W_ih + ((long)l * NDIM + srow) * 1024 + k4)
      : (W_hh + ((long)l * NDIM + srow) * 1024 + (k4 - 1024));
  float4 f = *(const float4*)src;
  uint64_t p = (uint64_t)f2bf(f.x) | ((uint64_t)f2bf(f.y) << 16) |
               ((uint64_t)f2bf(f.z) << 32) | ((uint64_t)f2bf(f.w) << 48);
  *(uint64_t*)(Wp + ((long)l * NDIM + c) * KDIM + k4) = p;
}

// ---- bias[c] = b_ih + b_hh, gate-interleaved.
__global__ void lstm_pack_bias(const float* __restrict__ b_ih, const float* __restrict__ b_hh,
                               float* __restrict__ bp) {
  int idx = blockIdx.x * blockDim.x + threadIdx.x;  // < NLAYERS*NDIM
  int c = idx & 4095, l = idx >> 12;
  int s = l * NDIM + (c & 3) * HID + (c >> 2);
  bp[idx] = b_ih[s] + b_hh[s];
}

// ---- fp32 [BATCH,HID] -> bf16 half of A (colOff = 0 for input, HID for h_prev).
__global__ void lstm_pack_half(const float* __restrict__ src, uint16_t* __restrict__ A, int colOff) {
  int idx = blockIdx.x * blockDim.x + threadIdx.x;  // < BATCH*HID/4
  int h4 = (idx & 255) << 2;
  int b  = idx >> 8;
  float4 f = *(const float4*)(src + (long)b * HID + h4);
  uint64_t p = (uint64_t)f2bf(f.x) | ((uint64_t)f2bf(f.y) << 16) |
               ((uint64_t)f2bf(f.z) << 32) | ((uint64_t)f2bf(f.w) << 48);
  *(uint64_t*)(A + (long)b * KDIM + colOff + h4) = p;
}

// ---- Fused GEMM + LSTM cell. One workgroup = 128x128 tile of gates.
__global__ __launch_bounds__(256)
void lstm_gemm_cell(const uint16_t* __restrict__ A,     // [BATCH][KDIM] bf16
                    const uint16_t* __restrict__ Wp,    // [NDIM][KDIM] bf16 (layer base)
                    const float*    __restrict__ bias,  // [NDIM] (layer base)
                    const float*    __restrict__ c_prev,// [BATCH][HID]
                    float*          __restrict__ out_h, // [BATCH][HID]
                    float*          __restrict__ out_c, // [BATCH][HID]
                    uint16_t*       __restrict__ A_next)// next layer A, write cols [0,HID)
{
  __shared__ __align__(16) unsigned char smem[2 * 2 * BM * LDT * 2]; // 40960 B
  uint16_t* sA = (uint16_t*)smem;              // [2][BM*LDT]
  uint16_t* sB = sA + 2 * BM * LDT;            // [2][BM*LDT], N-major
  float*    sG = (float*)smem;                 // epilogue: [64][132] (33792 B)

  const int tid  = threadIdx.x;
  const int lane = tid & 31;
  const int wid  = tid >> 5;
  const int wm   = wid >> 2;        // 0..1 -> 64 rows
  const int wn   = wid & 3;         // 0..3 -> 32 cols
  const int m0   = blockIdx.y * BM;
  const int n0   = blockIdx.x * BN;

  // global->reg staging map: thread covers rows lrow and lrow+64, 16B segment lseg
  const int lrow = tid >> 2;
  const int lseg = (tid & 3) * 8;
  const uint16_t* gA0 = A  + (long)(m0 + lrow) * KDIM + lseg;
  const uint16_t* gB0 = Wp + (long)(n0 + lrow) * KDIM + lseg;
  uint16_t* stA = sA + lrow * LDT + lseg;
  uint16_t* stB = sB + lrow * LDT + lseg;

  f32x8 acc[4][2];
#pragma unroll
  for (int t = 0; t < 4; ++t)
#pragma unroll
    for (int u = 0; u < 2; ++u)
#pragma unroll
      for (int r = 0; r < 8; ++r) acc[t][u][r] = 0.0f;

  // prologue: stage k-step 0
  {
    uint4 a0 = *(const uint4*)(gA0);
    uint4 a1 = *(const uint4*)(gA0 + 64 * KDIM);
    uint4 b0 = *(const uint4*)(gB0);
    uint4 b1 = *(const uint4*)(gB0 + 64 * KDIM);
    *(uint4*)(stA)            = a0;
    *(uint4*)(stA + 64 * LDT) = a1;
    *(uint4*)(stB)            = b0;
    *(uint4*)(stB + 64 * LDT) = b1;
  }
  __syncthreads();

  // fragment LDS base addresses (per-lane)
  const int arow = wm * 64 + (lane & 15);
  const int akc  = (lane >> 4) * 8;
  const int bcol = wn * 32 + (lane & 15);
  const int bkc  = (lane >> 4) * 8;

  // hoisted fragment registers: one stable allocation across all iterations
  FragU af[4], bf_[2];
  uint4 a0, a1, b0, b1;

#pragma unroll 1
  for (int it = 0; it < NKSTEP; ++it) {
    const bool more = (it + 1) < NKSTEP;
    if (more) {
      const int kb = (it + 1) * BK;
      a0 = *(const uint4*)(gA0 + kb);
      a1 = *(const uint4*)(gA0 + kb + 64 * KDIM);
      b0 = *(const uint4*)(gB0 + kb);
      b1 = *(const uint4*)(gB0 + kb + 64 * KDIM);
    }
    // pull the k+2 lines toward the WGP while WMMAs run (global_prefetch_b8)
    if (it + 2 < NKSTEP) {
      const int kp = (it + 2) * BK;
      __builtin_prefetch(gA0 + kp, 0, 0);
      __builtin_prefetch(gA0 + kp + 64 * KDIM, 0, 0);
      __builtin_prefetch(gB0 + kp, 0, 0);
      __builtin_prefetch(gB0 + kp + 64 * KDIM, 0, 0);
    }

    const int st = (it & 1) * (BM * LDT);
#pragma unroll
    for (int t = 0; t < 4; ++t) {
      const uint16_t* p = sA + st + (arow + t * 16) * LDT + akc;
      af[t].q[0] = *(const uint4*)(p);        // K = kc .. kc+7
      af[t].q[1] = *(const uint4*)(p + 16);   // K = kc+16 .. kc+23
    }
#pragma unroll
    for (int u = 0; u < 2; ++u) {
      const uint16_t* p = sB + st + (bcol + u * 16) * LDT + bkc;
      bf_[u].q[0] = *(const uint4*)(p);       // K = kc .. kc+7
      bf_[u].q[1] = *(const uint4*)(p + 16);  // K = kc+16 .. kc+23
    }
#pragma unroll
    for (int t = 0; t < 4; ++t)
#pragma unroll
      for (int u = 0; u < 2; ++u)
        acc[t][u] = __builtin_amdgcn_wmma_f32_16x16x32_bf16(
            false, af[t].v, false, bf_[u].v, (short)0, acc[t][u], false, false);

    if (more) {
      const int sn = ((it + 1) & 1) * (BM * LDT);
      *(uint4*)(stA + sn)            = a0;
      *(uint4*)(stA + sn + 64 * LDT) = a1;
      *(uint4*)(stB + sn)            = b0;
      *(uint4*)(stB + sn + 64 * LDT) = b1;
      __syncthreads();
    }
  }
  __syncthreads();

  // ---- fused epilogue: two 64-row phases through LDS, then LSTM cell math
#pragma unroll 1
  for (int ph = 0; ph < 2; ++ph) {
    if (wm == ph) {
#pragma unroll
      for (int t = 0; t < 4; ++t) {
        const int rowl = t * 16 + 8 * (lane >> 4);
#pragma unroll
        for (int u = 0; u < 2; ++u) {
          const int col = wn * 32 + u * 16 + (lane & 15);
#pragma unroll
          for (int r = 0; r < 8; ++r)
            sG[(rowl + r) * 132 + col] = acc[t][u][r];
        }
      }
    }
    __syncthreads();
#pragma unroll
    for (int j = 0; j < 8; ++j) {
      const int idx  = tid + 256 * j;       // 2048 (b,h) pairs per phase
      const int rowl = idx >> 5;
      const int hl   = idx & 31;
      const int b    = m0 + ph * 64 + rowl;
      const int hg   = (n0 >> 2) + hl;
      const float4 g4 = *(const float4*)&sG[rowl * 132 + hl * 4]; // i,f,g,o
      const float4 bb = *(const float4*)(bias + n0 + hl * 4);
      const float iv = sigf(g4.x + bb.x);
      const float fv = sigf(g4.y + bb.y);
      const float gv = tanhf(g4.z + bb.z);
      const float ov = sigf(g4.w + bb.w);
      const float cp = c_prev[(long)b * HID + hg];
      const float cn = fv * cp + iv * gv;
      const float hn = ov * tanhf(cn);
      out_h[(long)b * HID + hg] = hn;
      out_c[(long)b * HID + hg] = cn;
      A_next[(long)b * KDIM + hg] = f2bf(hn);
    }
    __syncthreads();
  }
}

extern "C" void kernel_launch(void* const* d_in, const int* in_sizes, int n_in,
                              void* d_out, int out_size, void* d_ws, size_t ws_size,
                              hipStream_t stream) {
  (void)in_sizes; (void)n_in; (void)out_size; (void)ws_size;
  const float* x    = (const float*)d_in[0];
  const float* h0   = (const float*)d_in[1];
  const float* c0   = (const float*)d_in[2];
  const float* W_ih = (const float*)d_in[3];
  const float* W_hh = (const float*)d_in[4];
  const float* b_ih = (const float*)d_in[5];
  const float* b_hh = (const float*)d_in[6];
  float* out = (float*)d_out;

  // workspace carve-up
  uint16_t* Wp = (uint16_t*)d_ws;                                     // 64 MB
  float*    bp = (float*)((char*)Wp + (size_t)NLAYERS * NDIM * KDIM * 2);
  uint16_t* A0 = (uint16_t*)((char*)bp + (size_t)NLAYERS * NDIM * 4); // 16 MB
  uint16_t* A1 = A0 + (size_t)BATCH * KDIM;                           // 16 MB

  lstm_pack_w   <<<(NLAYERS * NDIM * (KDIM / 4)) / 256, 256, 0, stream>>>(W_ih, W_hh, Wp);
  lstm_pack_bias<<<(NLAYERS * NDIM) / 256,              256, 0, stream>>>(b_ih, b_hh, bp);
  lstm_pack_half<<<(BATCH * HID / 4) / 256,             256, 0, stream>>>(x, A0, 0);

  uint16_t* Acur = A0; uint16_t* Anx = A1;
  for (int l = 0; l < NLAYERS; ++l) {
    lstm_pack_half<<<(BATCH * HID / 4) / 256, 256, 0, stream>>>(h0 + (size_t)l * BATCH * HID, Acur, HID);
    dim3 grid(NDIM / BN, BATCH / BM);
    lstm_gemm_cell<<<grid, 256, 0, stream>>>(
        Acur, Wp + (size_t)l * NDIM * KDIM, bp + (size_t)l * NDIM,
        c0 + (size_t)l * BATCH * HID,
        out + (size_t)l * BATCH * HID,
        out + (size_t)(NLAYERS + l) * BATCH * HID,
        Anx);
    uint16_t* t = Acur; Acur = Anx; Anx = t;
  }
}
